// Encoder_63685775065410
// MI455X (gfx1250) — matrix operand order repaired and checked by reading, not compiled, not run on hardware
//
#include <hip/hip_runtime.h>
#include <hip/hip_bf16.h>

// ---------------------------------------------------------------------------
// Transformer encoder (B=8,S=1024,H=1024,NH=16,HD=64,PF=4096,L=6) on gfx1250.
// All GEMMs + attention use v_wmma_f32_16x16x32_bf16 (bf16 in, f32 accum).
// GEMM LDS staging uses CDNA5 async global->LDS copies (ASYNCcnt) with
// double buffering so tile loads overlap the WMMA burst.
// ---------------------------------------------------------------------------

#define BB 8
#define SS 1024
#define HH 1024
#define NHH 16
#define HDD 64
#define PFF 4096
#define LL 6
#define MM (BB * SS)   // 8192 tokens

typedef __bf16 bf16;
typedef __bf16 v16bf __attribute__((ext_vector_type(16)));
typedef float  v8f   __attribute__((ext_vector_type(8)));

// ---- WMMA fragment helpers (layouts per CDNA5 ISA §7.12.2) ------
// A-matrix 16x32 bf16: lane(0..15)=row M, K = half*8 + 0..7 (v0..3) and
//                      16 + half*8 + 0..7 (v4..7), half = lane>>4.
__device__ __forceinline__ v16bf load_frag_a(const bf16* base, int ld) {
  const int lane = threadIdx.x & 31;
  const bf16* p = base + (size_t)(lane & 15) * ld + ((lane >> 4) * 8);
  v16bf f;
  uint4* fv = reinterpret_cast<uint4*>(&f);
  fv[0] = *reinterpret_cast<const uint4*>(p);
  fv[1] = *reinterpret_cast<const uint4*>(p + 16);
  return f;
}
// B-matrix 32x16 bf16 stored as Bt[n][k]: lane holds col n = lane&15,
// K = half*8 + 0..7 (v0..3) and 16 + half*8 + 0..7 (v4..7), same as A.
__device__ __forceinline__ v16bf load_frag_b(const bf16* base, int ld) {
  const int lane = threadIdx.x & 31;
  const bf16* p = base + (size_t)(lane & 15) * ld + ((lane >> 4) * 8);
  v16bf f;
  uint4* fv = reinterpret_cast<uint4*>(&f);
  fv[0] = *reinterpret_cast<const uint4*>(p);
  fv[1] = *reinterpret_cast<const uint4*>(p + 16);
  return f;
}
__device__ __forceinline__ v8f wmma_bf16(v16bf a, v16bf b, v8f c) {
  return __builtin_amdgcn_wmma_f32_16x16x32_bf16(false, a, false, b,
                                                 (short)0, c, false, false);
}

// ---- CDNA5 async global->LDS (CDNA5 ISA) -------------
// Per-lane: copies global bytes at gaddr into LDS at lds_off, same offset on both sides.
// Copies 32 contiguous bytes per lane (two b128 ops sharing one base).
__device__ __forceinline__ void async_copy32(unsigned int lds_off, const bf16* gptr) {
  unsigned long long ga = (unsigned long long)(uintptr_t)gptr;
  asm volatile("global_load_async_to_lds_b128 %0, %1, off\n\t"
               "global_load_async_to_lds_b128 %0, %1, off offset:16"
               :: "v"(lds_off), "v"(ga) : "memory");
}
__device__ __forceinline__ void wait_async0() {
  asm volatile("s_wait_asynccnt 0" ::: "memory");
}
// Generic LDS pointer -> 32-bit LDS byte address (flat aperture: addr[31:0]).
__device__ __forceinline__ unsigned int lds_addr32(const void* p) {
  return (unsigned int)(uintptr_t)p;
}

// ---- elementwise / layout kernels -----------------------------------------
__global__ __launch_bounds__(256)
void embed_kernel(const int* __restrict__ src, const float* __restrict__ tok_emb,
                  const float* __restrict__ pos_emb, float* __restrict__ xf,
                  bf16* __restrict__ xb) {
  size_t idx = (size_t)blockIdx.x * 256 + threadIdx.x;
  int hh = (int)(idx & (HH - 1));
  int s  = (int)((idx >> 10) & (SS - 1));
  int b  = (int)(idx >> 20);
  int tok = src[b * SS + s];
  float v = tok_emb[(size_t)tok * HH + hh] * 32.0f + pos_emb[(size_t)s * HH + hh];
  xf[idx] = v;
  xb[idx] = (bf16)v;
}

// fp32 W[K][N] -> bf16 Wt[N][K]
__global__ __launch_bounds__(256)
void convT_kernel(const float* __restrict__ W, bf16* __restrict__ Wt, int K, int N) {
  size_t idx = (size_t)blockIdx.x * 256 + threadIdx.x;
  int k = (int)(idx % K);
  int n = (int)(idx / K);
  Wt[idx] = (bf16)W[(size_t)k * N + n];
}

// V [B,S,H] bf16 -> Vt [B*NH, HD, S] bf16
__global__ __launch_bounds__(256)
void vtrans_kernel(const bf16* __restrict__ v, bf16* __restrict__ vt) {
  size_t idx = (size_t)blockIdx.x * 256 + threadIdx.x;
  int s  = (int)(idx & (SS - 1));
  int d  = (int)((idx >> 10) & (HDD - 1));
  int bh = (int)(idx >> 16);
  int b = bh >> 4, h = bh & 15;
  vt[idx] = v[(size_t)(b * SS + s) * HH + h * HDD + d];
}

// x = LayerNorm(res + delta) * g + b ; writes f32 and bf16 copies.
__global__ __launch_bounds__(256)
void ln_kernel(const float* __restrict__ res, const float* __restrict__ delta,
               const float* __restrict__ g, const float* __restrict__ beta,
               float* __restrict__ outF, bf16* __restrict__ outB) {
  __shared__ float red[256];
  const size_t base = (size_t)blockIdx.x * HH;
  float x[4];
  float s = 0.f;
  #pragma unroll
  for (int i = 0; i < 4; i++) {
    int c = threadIdx.x + i * 256;
    x[i] = res[base + c] + delta[base + c];
    s += x[i];
  }
  red[threadIdx.x] = s;
  __syncthreads();
  for (int o = 128; o > 0; o >>= 1) {
    if (threadIdx.x < o) red[threadIdx.x] += red[threadIdx.x + o];
    __syncthreads();
  }
  float mean = red[0] * (1.0f / HH);
  __syncthreads();
  float vs = 0.f;
  #pragma unroll
  for (int i = 0; i < 4; i++) { float d = x[i] - mean; vs += d * d; }
  red[threadIdx.x] = vs;
  __syncthreads();
  for (int o = 128; o > 0; o >>= 1) {
    if (threadIdx.x < o) red[threadIdx.x] += red[threadIdx.x + o];
    __syncthreads();
  }
  float rstd = rsqrtf(red[0] * (1.0f / HH) + 1e-5f);
  #pragma unroll
  for (int i = 0; i < 4; i++) {
    int c = threadIdx.x + i * 256;
    float y = (x[i] - mean) * rstd * g[c] + beta[c];
    outF[base + c] = y;
    outB[base + c] = (bf16)y;
  }
}

// ---- WMMA GEMM: C[M,N] = A[M,K](bf16) * Bt[N,K](bf16)^T + bias ------------
// Double-buffered LDS, async global->LDS staging, 128x128 block tile,
// 8 waves (2x4), each wave 64x32 = 8 accumulators, K stepped by 32.
// EPI: 0 -> f32 out, 1 -> bf16 out with ReLU, 2 -> bf16 out
template <int EPI>
__global__ __launch_bounds__(256, 2)
void gemm_kernel(const bf16* __restrict__ A, const bf16* __restrict__ Bt,
                 const float* __restrict__ bias, float* __restrict__ outF,
                 bf16* __restrict__ outB, int M, int N, int K) {
  __shared__ bf16 As[2][128][40];
  __shared__ bf16 Bs[2][128][40];
  const int tid  = threadIdx.x;
  const int lane = tid & 31;
  const int wave = tid >> 5;
  const int wm = wave >> 2, wn = wave & 3;      // 2 x 4 wave grid
  const int m0 = blockIdx.y * 128, n0 = blockIdx.x * 128;
  const int trow = tid >> 1, tcol = (tid & 1) * 16;

  // per-thread staging addresses (hoisted; 32B per thread per tile per matrix)
  const bf16* aSrc = A  + (size_t)(m0 + trow) * K + tcol;
  const bf16* bSrc = Bt + (size_t)(n0 + trow) * K + tcol;
  const unsigned int aDst[2] = { lds_addr32(&As[0][trow][tcol]),
                                 lds_addr32(&As[1][trow][tcol]) };
  const unsigned int bDst[2] = { lds_addr32(&Bs[0][trow][tcol]),
                                 lds_addr32(&Bs[1][trow][tcol]) };

  v8f acc[4][2];
  #pragma unroll
  for (int i = 0; i < 4; i++)
    #pragma unroll
    for (int j = 0; j < 2; j++)
      acc[i][j] = (v8f){0.f, 0.f, 0.f, 0.f, 0.f, 0.f, 0.f, 0.f};

  // prologue: stage tile 0
  async_copy32(aDst[0], aSrc);
  async_copy32(bDst[0], bSrc);
  wait_async0();
  __syncthreads();

  const int nt = K >> 5;
  for (int it = 0; it < nt; it++) {
    const int buf = it & 1;
    // prefetch next tile into the other buffer (overlaps WMMA below)
    if (it + 1 < nt) {
      async_copy32(aDst[buf ^ 1], aSrc + ((it + 1) << 5));
      async_copy32(bDst[buf ^ 1], bSrc + ((it + 1) << 5));
    }

    v16bf af[4], bfr[2];
    #pragma unroll
    for (int mi = 0; mi < 4; mi++)
      af[mi] = load_frag_a(&As[buf][wm * 64 + mi * 16][0], 40);
    #pragma unroll
    for (int ni = 0; ni < 2; ni++)
      bfr[ni] = load_frag_b(&Bs[buf][wn * 32 + ni * 16][0], 40);
    #pragma unroll
    for (int mi = 0; mi < 4; mi++)
      #pragma unroll
      for (int ni = 0; ni < 2; ni++)
        acc[mi][ni] = wmma_bf16(af[mi], bfr[ni], acc[mi][ni]);

    if (it + 1 < nt) wait_async0();  // own prefetch landed
    __syncthreads();                 // everyone done reading buf / writing buf^1
  }

  const int lh = lane >> 4, lo = lane & 15;
  #pragma unroll
  for (int mi = 0; mi < 4; mi++) {
    #pragma unroll
    for (int ni = 0; ni < 2; ni++) {
      int gcol = n0 + wn * 32 + ni * 16 + lo;
      float bb = bias[gcol];
      #pragma unroll
      for (int i = 0; i < 8; i++) {
        int grow = m0 + wm * 64 + mi * 16 + lh * 8 + i;
        float v = acc[mi][ni][i] + bb;
        if (EPI == 1) v = fmaxf(v, 0.f);
        if (EPI == 0) outF[(size_t)grow * N + gcol] = v;
        else          outB[(size_t)grow * N + gcol] = (bf16)v;
      }
    }
  }
}

// ---- fused flash attention: 1 wave = 16 query rows, 32-key tiles ----------
__global__ __launch_bounds__(128)
void flash_attn_kernel(const bf16* __restrict__ Q, const bf16* __restrict__ Kb,
                       const bf16* __restrict__ Vt,
                       const unsigned char* __restrict__ mask,
                       bf16* __restrict__ ctx) {
  __shared__ bf16 Plds[4][16][32];
  const int tid = threadIdx.x, lane = tid & 31, w = tid >> 5;
  const int bh = blockIdx.y, b = bh >> 4, h = bh & 15;
  const int q0 = blockIdx.x * 64 + w * 16;
  const int lh = lane >> 4, lo = lane & 15;

  const bf16* qbase = Q + (size_t)(b * SS + q0) * HH + h * HDD;
  const v16bf qa0 = load_frag_a(qbase, HH);       // hd 0..31
  const v16bf qa1 = load_frag_a(qbase + 32, HH);  // hd 32..63

  v8f acc[4];
  #pragma unroll
  for (int n = 0; n < 4; n++)
    acc[n] = (v8f){0.f, 0.f, 0.f, 0.f, 0.f, 0.f, 0.f, 0.f};
  float rmax[8], rsum[8];
  #pragma unroll
  for (int i = 0; i < 8; i++) { rmax[i] = -1e30f; rsum[i] = 0.f; }

  const v8f zf = (v8f){0.f, 0.f, 0.f, 0.f, 0.f, 0.f, 0.f, 0.f};
  for (int kt = 0; kt < SS; kt += 32) {
    // S = Q * K^T : K rows are natural B-frags (Bt[n][k] == K[key][hd])
    const bf16* kbase = Kb + (size_t)(b * SS + kt) * HH + h * HDD;
    v16bf k00 = load_frag_b(kbase, HH);
    v16bf k01 = load_frag_b(kbase + 32, HH);
    v16bf k10 = load_frag_b(kbase + (size_t)16 * HH, HH);
    v16bf k11 = load_frag_b(kbase + (size_t)16 * HH + 32, HH);
    v8f s0 = wmma_bf16(qa0, k00, zf);
    s0 = wmma_bf16(qa1, k01, s0);
    v8f s1 = wmma_bf16(qa0, k10, zf);
    s1 = wmma_bf16(qa1, k11, s1);

    const float mk0 = mask[b * SS + kt + lo]      ? 0.f : -1e10f;
    const float mk1 = mask[b * SS + kt + 16 + lo] ? 0.f : -1e10f;

    #pragma unroll
    for (int i = 0; i < 8; i++) {
      float a = s0[i] * 0.125f + mk0;   // 1/sqrt(64)
      float c = s1[i] * 0.125f + mk1;
      float mx = fmaxf(a, c);
      mx = fmaxf(mx, __shfl_xor(mx, 1, 32));
      mx = fmaxf(mx, __shfl_xor(mx, 2, 32));
      mx = fmaxf(mx, __shfl_xor(mx, 4, 32));
      mx = fmaxf(mx, __shfl_xor(mx, 8, 32));
      float nm = fmaxf(rmax[i], mx);
      float corr = __expf(rmax[i] - nm);
      float p0 = __expf(a - nm);
      float p1 = __expf(c - nm);
      float ts = p0 + p1;
      ts += __shfl_xor(ts, 1, 32);
      ts += __shfl_xor(ts, 2, 32);
      ts += __shfl_xor(ts, 4, 32);
      ts += __shfl_xor(ts, 8, 32);
      rsum[i] = rsum[i] * corr + ts;
      rmax[i] = nm;
      #pragma unroll
      for (int n = 0; n < 4; n++) acc[n][i] *= corr;
      // D-layout -> row-major P tile in LDS
      Plds[w][lh * 8 + i][lo]      = (bf16)p0;
      Plds[w][lh * 8 + i][16 + lo] = (bf16)p1;
    }
    __syncthreads();
    v16bf pa = load_frag_a(&Plds[w][0][0], 32);
    #pragma unroll
    for (int n = 0; n < 4; n++) {
      v16bf vf = load_frag_b(Vt + (size_t)(bh * HDD + n * 16) * SS + kt, SS);
      acc[n] = wmma_bf16(pa, vf, acc[n]);
    }
    __syncthreads();
  }

  #pragma unroll
  for (int i = 0; i < 8; i++) {
    float inv = 1.0f / rsum[i];
    int row = q0 + lh * 8 + i;
    #pragma unroll
    for (int n = 0; n < 4; n++) {
      int col = h * HDD + n * 16 + lo;
      ctx[(size_t)(b * SS + row) * HH + col] = (bf16)(acc[n][i] * inv);
    }
  }
}

// ---------------------------------------------------------------------------
extern "C" void kernel_launch(void* const* d_in, const int* in_sizes, int n_in,
                              void* d_out, int out_size, void* d_ws, size_t ws_size,
                              hipStream_t stream) {
  const int*   src      = (const int*)d_in[0];
  const unsigned char* src_mask = (const unsigned char*)d_in[1];
  const float* tok_emb  = (const float*)d_in[2];
  const float* pos_emb  = (const float*)d_in[3];
  const float* Wq = (const float*)d_in[4];  const float* bq = (const float*)d_in[5];
  const float* Wk = (const float*)d_in[6];  const float* bk = (const float*)d_in[7];
  const float* Wv = (const float*)d_in[8];  const float* bv = (const float*)d_in[9];
  const float* Wo = (const float*)d_in[10]; const float* bo = (const float*)d_in[11];
  const float* W1 = (const float*)d_in[12]; const float* b1 = (const float*)d_in[13];
  const float* W2 = (const float*)d_in[14]; const float* b2 = (const float*)d_in[15];
  const float* ln1_g = (const float*)d_in[16]; const float* ln1_b = (const float*)d_in[17];
  const float* ln2_g = (const float*)d_in[18]; const float* ln2_b = (const float*)d_in[19];
  float* out = (float*)d_out;

  // Workspace carve-out (~248 MiB total)
  char* ws = (char*)d_ws;
  size_t off = 0;
  auto carve = [&](size_t bytes) {
    void* p = ws + off;
    off += (bytes + 255) & ~(size_t)255;
    return p;
  };
  float* xf   = (float*)carve((size_t)MM * HH * 4);
  bf16*  xb   = (bf16*)carve((size_t)MM * HH * 2);
  bf16*  qb   = (bf16*)carve((size_t)MM * HH * 2);
  bf16*  kb   = (bf16*)carve((size_t)MM * HH * 2);
  bf16*  vb   = (bf16*)carve((size_t)MM * HH * 2);
  bf16*  vt   = (bf16*)carve((size_t)MM * HH * 2);
  bf16*  ctx  = (bf16*)carve((size_t)MM * HH * 2);
  float* tmpf = (float*)carve((size_t)MM * HH * 4);
  bf16*  h1b  = (bf16*)carve((size_t)MM * PFF * 2);
  bf16*  wqb  = (bf16*)carve((size_t)HH * HH * 2);
  bf16*  wkb  = (bf16*)carve((size_t)HH * HH * 2);
  bf16*  wvb  = (bf16*)carve((size_t)HH * HH * 2);
  bf16*  wob  = (bf16*)carve((size_t)HH * HH * 2);
  bf16*  w1b  = (bf16*)carve((size_t)HH * PFF * 2);
  bf16*  w2b  = (bf16*)carve((size_t)PFF * HH * 2);
  (void)ws_size; (void)n_in; (void)in_sizes; (void)out_size;

  const dim3 blk256(256), blk128(128);
  const dim3 gEmb((size_t)MM * HH / 256);
  const dim3 gCwHH((size_t)HH * HH / 256);
  const dim3 gCwHP((size_t)HH * PFF / 256);
  const dim3 gGemmH(HH / 128, MM / 128);   // N=1024 tiles
  const dim3 gGemmP(PFF / 128, MM / 128);  // N=4096 tiles
  const dim3 gVt((size_t)MM * HH / 256);
  const dim3 gFlash(SS / 64, BB * NHH);
  const dim3 gLN(MM);

  // x = tok_emb[src]*sqrt(H) + pos_emb
  embed_kernel<<<gEmb, blk256, 0, stream>>>(src, tok_emb, pos_emb, xf, xb);

  for (int l = 0; l < LL; l++) {
    const size_t wHH = (size_t)l * HH * HH;
    const size_t wHP = (size_t)l * HH * PFF;
    convT_kernel<<<gCwHH, blk256, 0, stream>>>(Wq + wHH, wqb, HH, HH);
    convT_kernel<<<gCwHH, blk256, 0, stream>>>(Wk + wHH, wkb, HH, HH);
    convT_kernel<<<gCwHH, blk256, 0, stream>>>(Wv + wHH, wvb, HH, HH);
    convT_kernel<<<gCwHH, blk256, 0, stream>>>(Wo + wHH, wob, HH, HH);
    convT_kernel<<<gCwHP, blk256, 0, stream>>>(W1 + wHP, w1b, HH, PFF);
    convT_kernel<<<gCwHP, blk256, 0, stream>>>(W2 + wHP, w2b, PFF, HH);

    // Q,K,V projections (bf16 out)
    gemm_kernel<2><<<gGemmH, blk256, 0, stream>>>(xb, wqb, bq + (size_t)l * HH,
                                                  nullptr, qb, MM, HH, HH);
    gemm_kernel<2><<<gGemmH, blk256, 0, stream>>>(xb, wkb, bk + (size_t)l * HH,
                                                  nullptr, kb, MM, HH, HH);
    gemm_kernel<2><<<gGemmH, blk256, 0, stream>>>(xb, wvb, bv + (size_t)l * HH,
                                                  nullptr, vb, MM, HH, HH);
    vtrans_kernel<<<gVt, blk256, 0, stream>>>(vb, vt);

    // fused softmax(QK^T/sqrt(d)) V
    flash_attn_kernel<<<gFlash, blk128, 0, stream>>>(qb, kb, vt, src_mask, ctx);

    // output projection (f32) + residual + LN1
    gemm_kernel<0><<<gGemmH, blk256, 0, stream>>>(ctx, wob, bo + (size_t)l * HH,
                                                  tmpf, nullptr, MM, HH, HH);
    ln_kernel<<<gLN, blk256, 0, stream>>>(xf, tmpf, ln1_g + (size_t)l * HH,
                                          ln1_b + (size_t)l * HH, xf, xb);

    // FFN: relu(x W1 + b1) W2 + b2, residual + LN2
    gemm_kernel<1><<<gGemmP, blk256, 0, stream>>>(xb, w1b, b1 + (size_t)l * PFF,
                                                  nullptr, h1b, MM, PFF, HH);
    gemm_kernel<0><<<gGemmH, blk256, 0, stream>>>(h1b, w2b, b2 + (size_t)l * HH,
                                                  tmpf, nullptr, MM, HH, PFF);
    float* lnOut = (l == LL - 1) ? out : xf;
    ln_kernel<<<gLN, blk256, 0, stream>>>(xf, tmpf, ln2_g + (size_t)l * HH,
                                          ln2_b + (size_t)l * HH, lnOut, xb);
  }
}
